// TrajectoryModeling_81836306858612
// MI455X (gfx1250) — compile-verified, matrix operand order checked by reading, not run on hardware
//
#include <hip/hip_runtime.h>
#include <hip/hip_bf16.h>

// ---------------------------------------------------------------------------
// Types
// ---------------------------------------------------------------------------
typedef __bf16 bf16_t;
typedef bf16_t v16bf __attribute__((ext_vector_type(16)));
typedef float  v8f   __attribute__((ext_vector_type(8)));
typedef unsigned int u32x4 __attribute__((ext_vector_type(4)));
typedef float  f32x4 __attribute__((ext_vector_type(4)));

// ---------------------------------------------------------------------------
// Problem dimensions (fixed by the reference)
// ---------------------------------------------------------------------------
constexpr int B_ = 2, G_ = 4, T_ = 64, N_ = 300, D_ = 256, MD_ = 32;
constexpr long long TOK = (long long)B_ * G_ * T_ * N_;   // 153600 tokens
constexpr int TILE_BLOCKS = (int)(TOK / 16);              // 9600 (TOK % 16 == 0)

// packed bf16 weight offsets inside d_ws (in bf16 elements)
constexpr size_t OFF_ADP1 = 0;                               // 256x1024
constexpr size_t OFF_ADP2 = OFF_ADP1 + (size_t)256 * 1024;   // 1024x256
constexpr size_t OFF_FUS1 = OFF_ADP2 + (size_t)1024 * 256;   // 288x256
constexpr size_t OFF_FUS2 = OFF_FUS1 + (size_t)288 * 256;    // 256x256
constexpr size_t OFF_FFN1 = OFF_FUS2 + (size_t)256 * 256;    // 256x1024
constexpr size_t OFF_FFN2 = OFF_FFN1 + (size_t)256 * 1024;   // 1024x256

// LDS row strides (bf16 strides multiple of 8 -> 16B-aligned b128 frag loads)
constexpr int SA = 296;   // bf16 stride, holds up to 288 cols (cat = x | motion)
constexpr int SH = 1032;  // bf16 stride, holds 1024 hidden cols
constexpr int SC = 260;   // f32 stride, holds 256 cols (padded vs bank conflicts)

union F16x { u32x4 u4[2]; v16bf v; };

__device__ __forceinline__ float gelu_f(float x) {
  return 0.5f * x * (1.0f + erff(x * 0.70710678118654752440f));
}

__device__ __forceinline__ v8f vzero8() {
  v8f z;
  #pragma unroll
  for (int i = 0; i < 8; ++i) z[i] = 0.0f;
  return z;
}

// ---------------------------------------------------------------------------
// WMMA fragment loads (layouts per CDNA5 ISA, 16-bit 16x32 tiles)
// A (row-major in LDS): lane<16 -> row=lane, k-local {0..7, 16..23};
//                       lane>=16 -> row=lane-16, k-local {8..15, 24..31}
// B (pre-packed in d_ws by pack_weight_kernel): 32 lanes x 16 bf16 contiguous
// ---------------------------------------------------------------------------
__device__ __forceinline__ v16bf load_a_frag(const bf16_t* base, int stride,
                                             int kt, int lane) {
  const int m  = lane & 15;
  const int kb = (kt << 5) + ((lane >> 4) << 3);
  const bf16_t* p = base + m * stride + kb;
  F16x f;
  f.u4[0] = *(const u32x4*)(p);        // k-local +0..7   (16B aligned)
  f.u4[1] = *(const u32x4*)(p + 16);   // k-local +16..23 (16B aligned)
  return f.v;
}

__device__ __forceinline__ v16bf load_b_frag(const bf16_t* wp, int KT,
                                             int nt, int kt, int lane) {
  const bf16_t* p = wp + ((((size_t)nt * KT + kt) << 5) + lane) * 16;
  F16x f;
  f.u4[0] = *(const u32x4*)(p);
  f.u4[1] = *(const u32x4*)(p + 8);
  return f.v;
}

// Accumulate a group of 4 adjacent 16-wide N-tiles over K = KT*32
__device__ __forceinline__ void gemm_group4(const bf16_t* A, int As,
                                            const bf16_t* Wp, int KT,
                                            int nt0, int lane, v8f acc[4]) {
  for (int kt = 0; kt < KT; ++kt) {
    v16bf a = load_a_frag(A, As, kt, lane);
    #pragma unroll
    for (int i = 0; i < 4; ++i) {
      v16bf b = load_b_frag(Wp, KT, nt0 + i, kt, lane);
      acc[i] = __builtin_amdgcn_wmma_f32_16x16x32_bf16(
          /*neg_a=*/false, a, /*neg_b=*/false, b,
          /*c_mod=*/(short)0, acc[i], /*reuse_a=*/false, /*reuse_b=*/false);
    }
  }
}

// ---------------------------------------------------------------------------
// Kernel 0: pack a f32 weight matrix [K x N] (row-major) into bf16 fragments
// ---------------------------------------------------------------------------
__global__ void pack_weight_kernel(const float* __restrict__ src,
                                   bf16_t* __restrict__ dst, int K, int N) {
  const int t = blockIdx.x * blockDim.x + threadIdx.x;
  const int KT = K >> 5;
  const int total = (N >> 4) * KT * 32;
  if (t >= total) return;
  const int lane = t & 31;
  const int tile = t >> 5;
  const int kt = tile % KT;
  const int nt = tile / KT;
  const int n  = (nt << 4) + (lane & 15);
  const int kb = (kt << 5) + ((lane >> 4) << 3);
  bf16_t* o = dst + ((size_t)t << 4);
  #pragma unroll
  for (int j = 0; j < 8; ++j) o[j]     = (bf16_t)src[(size_t)(kb + j) * N + n];
  #pragma unroll
  for (int j = 0; j < 8; ++j) o[8 + j] = (bf16_t)src[(size_t)(kb + 16 + j) * N + n];
}

// ---------------------------------------------------------------------------
// Kernel 1: motion features. One 64-thread block per trajectory (b,g,n).
// ---------------------------------------------------------------------------
__global__ void __launch_bounds__(64)
motion_kernel(const float* __restrict__ boxes, const unsigned char* __restrict__ masks,
              const int* __restrict__ times,
              const float* __restrict__ mw1, const float* __restrict__ mb1,
              const float* __restrict__ mw2, const float* __restrict__ mb2,
              float* __restrict__ motion) {
  __shared__ int   sKey[T_];
  __shared__ int   sTime[T_];
  __shared__ float sBox[T_][4];
  __shared__ float sEnc[T_][MD_];
  __shared__ int   sWr[T_];
  __shared__ float sW1[12 * 16];
  __shared__ float sB1[16];
  __shared__ float sW2[16 * MD_];
  __shared__ float sB2[MD_];

  const int t  = threadIdx.x;                 // frame slot 0..63
  const int n  = blockIdx.x % N_;
  const int bg = blockIdx.x / N_;             // b*G + g
  const size_t idx = ((size_t)bg * T_ + t) * N_ + n;

  for (int i = t; i < 12 * 16; i += T_) sW1[i] = mw1[i];
  if (t < 16) sB1[t] = mb1[t];
  for (int i = t; i < 16 * MD_; i += T_) sW2[i] = mw2[i];
  if (t < MD_) sB2[t] = mb2[t];
  sWr[t] = 0;

  const int  tm    = times[idx];
  const bool valid = (masks[idx] == 0);
  const int  key   = valid ? tm : (T_ + 1);
  sKey[t] = key;
  __syncthreads();

  int rank = 0, cnt = 0;                      // stable rank == argsort position
  for (int j = 0; j < T_; ++j) {
    const int kj = sKey[j];
    if ((kj < key) || (kj == key && j < t)) ++rank;
    if (kj < T_ + 1) ++cnt;
  }
  sTime[rank] = tm;
  const f32x4 bx = *(const f32x4*)(boxes + idx * 4);
  sBox[rank][0] = bx.x; sBox[rank][1] = bx.y; sBox[rank][2] = bx.z; sBox[rank][3] = bx.w;
  __syncthreads();

  const int p   = t;
  const int pm1 = (p >= 1) ? p - 1 : 0;
  const int pm2 = (p >= 2) ? p - 2 : 0;
  const int stp = sTime[p], stm1 = sTime[pm1], stm2 = sTime[pm2];
  const bool inc = (p < cnt);
  const int d1 = stp - stm1, d2 = stm1 - stm2;
  const float dtp  = (float)(d1 > 1 ? d1 : 1);
  const float dtm1 = (float)(d2 > 1 ? d2 : 1);

  float mv[12];
  #pragma unroll
  for (int c = 0; c < 4; ++c) {
    const float b0  = sBox[p][c];
    const float bm1 = sBox[pm1][c];
    const float bm2 = sBox[pm2][c];
    const float vel   = (p >= 1 && inc) ? (b0 - bm1) / dtp : 0.0f;
    const float velm1 = (p >= 2 && (p - 1) < cnt) ? (bm1 - bm2) / dtm1 : 0.0f;
    const float acc   = (p >= 2 && inc) ? (vel - velm1) / dtp : 0.0f;
    mv[c] = b0; mv[4 + c] = vel; mv[8 + c] = acc;
  }

  float h[16];
  #pragma unroll
  for (int o = 0; o < 16; ++o) {
    float a = sB1[o];
    #pragma unroll
    for (int i = 0; i < 12; ++i) a += mv[i] * sW1[i * 16 + o];
    h[o] = gelu_f(a);
  }

  const bool wr = inc && (cnt >= 2) && (stp < T_);
  if (wr) {
    for (int o = 0; o < MD_; ++o) {
      float a = sB2[o];
      #pragma unroll
      for (int i = 0; i < 16; ++i) a += h[i] * sW2[i * MD_ + o];
      sEnc[stp][o] = a;
    }
    sWr[stp] = 1;
  }
  __syncthreads();

  float* orow = motion + idx * MD_;
  if (sWr[t]) {
    for (int c = 0; c < MD_; ++c) orow[c] = sEnc[t][c];
  } else {
    for (int c = 0; c < MD_; ++c) orow[c] = 0.0f;
  }
}

// ---------------------------------------------------------------------------
// LN row statistics over 256 cols of sC (128 threads: 8 per row of 16 rows)
// ---------------------------------------------------------------------------
__device__ __forceinline__ void row_stats_256(const float* sC_, float* sRed_,
                                              float* sMean_, float* sRstd_, int tid) {
  const int row = tid >> 3, sub = tid & 7;
  float s = 0.0f, s2 = 0.0f;
  const float* rp = sC_ + row * SC + sub * 32;
  #pragma unroll 8
  for (int c = 0; c < 32; ++c) { const float v = rp[c]; s += v; s2 += v * v; }
  sRed_[(row * 8 + sub) * 2 + 0] = s;
  sRed_[(row * 8 + sub) * 2 + 1] = s2;
  __syncthreads();
  if (sub == 0) {
    float S = 0.0f, S2 = 0.0f;
    #pragma unroll
    for (int j = 0; j < 8; ++j) {
      S  += sRed_[(row * 8 + j) * 2 + 0];
      S2 += sRed_[(row * 8 + j) * 2 + 1];
    }
    const float m   = S  * (1.0f / 256.0f);
    const float var = S2 * (1.0f / 256.0f) - m * m;
    sMean_[row] = m;
    sRstd_[row] = rsqrtf(var + 1e-5f);
  }
  __syncthreads();
}

// ---------------------------------------------------------------------------
// Kernel 2: whole per-token pipeline, 16 tokens per block, 4 waves (128 thr).
// ---------------------------------------------------------------------------
__global__ void __launch_bounds__(128)
traj_fused_kernel(const float* __restrict__ feats, const float* __restrict__ motion,
                  const bf16_t* __restrict__ ws,
                  const float* __restrict__ adp_b1, const float* __restrict__ adp_b2,
                  const float* __restrict__ norm_g, const float* __restrict__ norm_b,
                  const float* __restrict__ fus_b1, const float* __restrict__ fus_b2,
                  const float* __restrict__ ffn_b1, const float* __restrict__ ffn_b2,
                  const float* __restrict__ fnorm_g, const float* __restrict__ fnorm_b,
                  float* __restrict__ out) {
  __shared__ bf16_t sA[16 * SA];     // A / cat / F1 / fused0(bf16)
  __shared__ bf16_t sH[16 * SH];     // 1024-wide hidden activations
  __shared__ float  sC[16 * SC];     // f32 scratch (pre-LN rows, residual)
  __shared__ float  sRed[16 * 8 * 2];
  __shared__ float  sMean[16];
  __shared__ float  sRstd[16];

  const int tid  = threadIdx.x;
  const int lane = tid & 31;
  const int w    = tid >> 5;                      // wave 0..3
  const int rb   = (lane >> 4) << 3;              // C/D row base (0 or 8)
  const size_t tok0 = (size_t)blockIdx.x * 16;

  const bf16_t* adp_w1p = ws + OFF_ADP1;
  const bf16_t* adp_w2p = ws + OFF_ADP2;
  const bf16_t* fus_w1p = ws + OFF_FUS1;
  const bf16_t* fus_w2p = ws + OFF_FUS2;
  const bf16_t* ffn_w1p = ws + OFF_FFN1;
  const bf16_t* ffn_w2p = ws + OFF_FFN2;

  // ---- load feature tile [16 x 256] f32 -> bf16 in sA ----
  {
    const f32x4* f4 = (const f32x4*)(feats + tok0 * D_);
    for (int i = tid; i < 16 * (D_ / 4); i += 128) {
      const f32x4 v = f4[i];
      const int row = i >> 6;
      const int col = (i & 63) << 2;
      bf16_t* p = sA + row * SA + col;
      p[0] = (bf16_t)v.x; p[1] = (bf16_t)v.y; p[2] = (bf16_t)v.z; p[3] = (bf16_t)v.w;
    }
  }
  __syncthreads();

  // ---- GEMM1: feats @ adp_w1 (K=256 -> N=1024), gelu -> sH (bf16) ----
  for (int g = 0; g < 4; ++g) {
    const int nt0 = w * 16 + g * 4;
    v8f acc[4] = {vzero8(), vzero8(), vzero8(), vzero8()};
    gemm_group4(sA, SA, adp_w1p, 8, nt0, lane, acc);
    #pragma unroll
    for (int i = 0; i < 4; ++i) {
      const int col = ((nt0 + i) << 4) + (lane & 15);
      const float bias = adp_b1[col];
      #pragma unroll
      for (int r = 0; r < 8; ++r)
        sH[(rb + r) * SH + col] = (bf16_t)gelu_f(acc[i][r] + bias);
    }
  }
  __syncthreads();

  // ---- GEMM2: h @ adp_w2 (K=1024 -> N=256), + bias + feats residual -> sC ----
  {
    const int nt0 = w * 4;
    v8f acc[4] = {vzero8(), vzero8(), vzero8(), vzero8()};
    gemm_group4(sH, SH, adp_w2p, 32, nt0, lane, acc);
    #pragma unroll
    for (int i = 0; i < 4; ++i) {
      const int col = ((nt0 + i) << 4) + (lane & 15);
      const float bias = adp_b2[col];
      #pragma unroll
      for (int r = 0; r < 8; ++r) {
        const int row = rb + r;
        sC[row * SC + col] = acc[i][r] + bias + feats[(tok0 + row) * D_ + col];
      }
    }
  }
  __syncthreads();

  // ---- LayerNorm 1 -> x (bf16) into sA cols 0..255 ----
  row_stats_256(sC, sRed, sMean, sRstd, tid);
  {
    const int row = tid >> 3, sub = tid & 7;
    const float m = sMean[row], rs = sRstd[row];
    for (int c = sub * 32; c < sub * 32 + 32; ++c)
      sA[row * SA + c] = (bf16_t)((sC[row * SC + c] - m) * rs * norm_g[c] + norm_b[c]);
  }
  // ---- append motion (bf16) into sA cols 256..287 ----
  for (int i = tid; i < 16 * MD_; i += 128) {
    const int row = i >> 5, c = i & 31;
    sA[row * SA + 256 + c] = (bf16_t)motion[(tok0 + row) * MD_ + c];
  }
  __syncthreads();

  // ---- GEMM3: cat @ fus_w1 (K=288 -> N=256), gelu -> sA (in place) ----
  {
    const int nt0 = w * 4;
    v8f acc[4] = {vzero8(), vzero8(), vzero8(), vzero8()};
    gemm_group4(sA, SA, fus_w1p, 9, nt0, lane, acc);
    __syncthreads();                       // all waves done reading cat
    #pragma unroll
    for (int i = 0; i < 4; ++i) {
      const int col = ((nt0 + i) << 4) + (lane & 15);
      const float bias = fus_b1[col];
      #pragma unroll
      for (int r = 0; r < 8; ++r)
        sA[(rb + r) * SA + col] = (bf16_t)gelu_f(acc[i][r] + bias);
    }
    __syncthreads();
  }

  // ---- GEMM4: f1 @ fus_w2 (K=256 -> N=256) -> fused0 (sC f32 + sA bf16) ----
  {
    const int nt0 = w * 4;
    v8f acc[4] = {vzero8(), vzero8(), vzero8(), vzero8()};
    gemm_group4(sA, SA, fus_w2p, 8, nt0, lane, acc);
    __syncthreads();
    #pragma unroll
    for (int i = 0; i < 4; ++i) {
      const int col = ((nt0 + i) << 4) + (lane & 15);
      const float bias = fus_b2[col];
      #pragma unroll
      for (int r = 0; r < 8; ++r) {
        const int row = rb + r;
        const float f0 = acc[i][r] + bias;
        sC[row * SC + col] = f0;
        sA[row * SA + col] = (bf16_t)f0;
      }
    }
    __syncthreads();
  }

  // ---- GEMM5: fused0 @ ffn_w1 (K=256 -> N=1024), gelu -> sH ----
  for (int g = 0; g < 4; ++g) {
    const int nt0 = w * 16 + g * 4;
    v8f acc[4] = {vzero8(), vzero8(), vzero8(), vzero8()};
    gemm_group4(sA, SA, ffn_w1p, 8, nt0, lane, acc);
    #pragma unroll
    for (int i = 0; i < 4; ++i) {
      const int col = ((nt0 + i) << 4) + (lane & 15);
      const float bias = ffn_b1[col];
      #pragma unroll
      for (int r = 0; r < 8; ++r)
        sH[(rb + r) * SH + col] = (bf16_t)gelu_f(acc[i][r] + bias);
    }
  }
  __syncthreads();

  // ---- GEMM6: h2 @ ffn_w2 (K=1024 -> N=256), + bias + fused0 residual ----
  {
    const int nt0 = w * 4;
    v8f acc[4] = {vzero8(), vzero8(), vzero8(), vzero8()};
    gemm_group4(sH, SH, ffn_w2p, 32, nt0, lane, acc);
    #pragma unroll
    for (int i = 0; i < 4; ++i) {
      const int col = ((nt0 + i) << 4) + (lane & 15);
      const float bias = ffn_b2[col];
      #pragma unroll
      for (int r = 0; r < 8; ++r) {
        const int row = rb + r;
        sC[row * SC + col] = acc[i][r] + bias + sC[row * SC + col];
      }
    }
  }
  __syncthreads();

  // ---- LayerNorm 2 -> out (f32) ----
  row_stats_256(sC, sRed, sMean, sRstd, tid);
  {
    const int row = tid >> 3, sub = tid & 7;
    const float m = sMean[row], rs = sRstd[row];
    for (int c = sub * 32; c < sub * 32 + 32; ++c)
      out[(tok0 + row) * D_ + c] = (sC[row * SC + c] - m) * rs * fnorm_g[c] + fnorm_b[c];
  }
}

// ---------------------------------------------------------------------------
// Host launcher
// ---------------------------------------------------------------------------
extern "C" void kernel_launch(void* const* d_in, const int* in_sizes, int n_in,
                              void* d_out, int out_size, void* d_ws, size_t ws_size,
                              hipStream_t stream) {
  (void)in_sizes; (void)n_in; (void)out_size; (void)ws_size;

  const float* feats   = (const float*)d_in[0];
  const float* boxes   = (const float*)d_in[1];
  const unsigned char* masks = (const unsigned char*)d_in[2];  // bool mask
  const int*   times   = (const int*)d_in[3];
  const float* adp_w1  = (const float*)d_in[4];
  const float* adp_b1  = (const float*)d_in[5];
  const float* adp_w2  = (const float*)d_in[6];
  const float* adp_b2  = (const float*)d_in[7];
  const float* norm_g  = (const float*)d_in[8];
  const float* norm_b  = (const float*)d_in[9];
  const float* mot_w1  = (const float*)d_in[10];
  const float* mot_b1  = (const float*)d_in[11];
  const float* mot_w2  = (const float*)d_in[12];
  const float* mot_b2  = (const float*)d_in[13];
  const float* fus_w1  = (const float*)d_in[14];
  const float* fus_b1  = (const float*)d_in[15];
  const float* fus_w2  = (const float*)d_in[16];
  const float* fus_b2  = (const float*)d_in[17];
  const float* ffn_w1  = (const float*)d_in[18];
  const float* ffn_b1  = (const float*)d_in[19];
  const float* ffn_w2  = (const float*)d_in[20];
  const float* ffn_b2  = (const float*)d_in[21];
  const float* fnorm_g = (const float*)d_in[22];
  const float* fnorm_b = (const float*)d_in[23];

  float* out_fused  = (float*)d_out;
  float* out_motion = out_fused + (size_t)TOK * D_;   // second tuple element
  bf16_t* ws = (bf16_t*)d_ws;

  // pack weights (f32 -> bf16, WMMA B-fragment layout) into workspace
  auto pack = [&](const float* src, size_t off, int K, int N) {
    const int total = (N / 16) * (K / 32) * 32;
    pack_weight_kernel<<<(total + 255) / 256, 256, 0, stream>>>(src, ws + off, K, N);
  };
  pack(adp_w1, OFF_ADP1, 256, 1024);
  pack(adp_w2, OFF_ADP2, 1024, 256);
  pack(fus_w1, OFF_FUS1, 288, 256);
  pack(fus_w2, OFF_FUS2, 256, 256);
  pack(ffn_w1, OFF_FFN1, 256, 1024);
  pack(ffn_w2, OFF_FFN2, 1024, 256);

  // motion branch: one block per trajectory
  motion_kernel<<<B_ * G_ * N_, T_, 0, stream>>>(boxes, masks, times,
                                                 mot_w1, mot_b1, mot_w2, mot_b2,
                                                 out_motion);

  // main fused per-token pipeline
  traj_fused_kernel<<<TILE_BLOCKS, 128, 0, stream>>>(feats, out_motion, ws,
      adp_b1, adp_b2, norm_g, norm_b, fus_b1, fus_b2,
      ffn_b1, ffn_b2, fnorm_g, fnorm_b, out_fused);
}
